// Encoder_44985487458613
// MI455X (gfx1250) — matrix operand order repaired and checked by reading, not compiled, not run on hardware
//
#include <hip/hip_runtime.h>
#include <hip/hip_bf16.h>
#include <math.h>

// ---------------------------------------------------------------------------
// Problem constants (match the reference)
// ---------------------------------------------------------------------------
#define NN   100000
#define EE   1600000
#define BB   128
#define DIN  32
#define DE   16
#define DEMB 64
#define HH   4
#define CC   16
#define LL   2
#define DGIN 96
#define EF   (EE + NN)   // edges + self loops

#define CDIV(a, b) (((a) + (b) - 1) / (b))

typedef __bf16 v16bf __attribute__((ext_vector_type(16)));
typedef float  v8f   __attribute__((ext_vector_type(8)));
typedef float  v4f   __attribute__((ext_vector_type(4)));

// ---------------------------------------------------------------------------
// Helpers
// ---------------------------------------------------------------------------
__device__ __forceinline__ void atomicMaxFloat(float* addr, float val) {
    // classic monotone punning: works across mixed signs
    if (val >= 0.0f) atomicMax((int*)addr, __float_as_int(val));
    else             atomicMin((unsigned int*)addr, __float_as_uint(val));
}

__global__ __launch_bounds__(256) void fill_kernel(float* p, size_t n, float v) {
    size_t i = (size_t)blockIdx.x * blockDim.x + threadIdx.x;
    if (i < n) p[i] = v;
}

// ---------------------------------------------------------------------------
// Self-loop attrs: deg[d] += 1 ; loop_sum[d] += eattr[e]
// ---------------------------------------------------------------------------
__global__ __launch_bounds__(256) void deg_loop_kernel(
    const int* __restrict__ dst, const float* __restrict__ eattr,
    float* __restrict__ deg, float* __restrict__ loop_sum) {
    int e = blockIdx.x * blockDim.x + threadIdx.x;
    if (e >= EE) return;
    int d = dst[e];
    atomicAdd(&deg[d], 1.0f);
    const float* row = eattr + (size_t)e * DE;
#pragma unroll
    for (int j = 0; j < DE; ++j) atomicAdd(&loop_sum[(size_t)d * DE + j], row[j]);
}

__global__ __launch_bounds__(256) void loop_div_kernel(
    float* __restrict__ loop_attr, const float* __restrict__ deg) {
    size_t i = (size_t)blockIdx.x * blockDim.x + threadIdx.x;
    if (i >= (size_t)NN * DE) return;
    loop_attr[i] /= fmaxf(deg[i >> 4], 1.0f);
}

// ---------------------------------------------------------------------------
// Templated WMMA GEMM:
//   Out[N,64] = act( concat(A1[:,W1], A2[:,W2]) @ Wt[K,64] + bias )
// bf16 inputs, f32 accumulate via v_wmma_f32_16x16x32_bf16 (wave32).
// One wave -> one 16-row tile, 4 col tiles.  K = W1+W2 in {32,64,96}; every
// 32-wide k-block lies wholly inside one concat source (W1,W2 are 32-mults).
// Weights are staged in LDS pre-swizzled into B-fragment order so each lane
// reads its v16bf fragment with two ds_load_b128 (no scalar LDS traffic).
// ---------------------------------------------------------------------------
template <int W1, int W2, int RELU>
__global__ __launch_bounds__(256) void wmma_gemm_t(
    const float* __restrict__ A1, const float* __restrict__ A2,
    const float* __restrict__ Wt, const float* __restrict__ bias,
    float* __restrict__ Out, int nrows)
{
    constexpr int K  = W1 + W2;
    constexpr int KB = K / 32;
    __shared__ v16bf Wsw[KB * 4 * 32];           // <= 12 KB, B-fragment order

    // cooperative fill: one whole fragment per iteration per thread
    for (int f = threadIdx.x; f < KB * 4 * 32; f += blockDim.x) {
        const int lane_f = f & 31;
        const int ct     = (f >> 5) & 3;
        const int kbb    = f >> 7;
        const int col    = ct * 16 + (lane_f & 15);
        const int k0     = kbb * 32 + ((lane_f >> 4) ? 8 : 0);
        v16bf frag;
#pragma unroll
        for (int i = 0; i < 16; ++i) frag[i] = (__bf16)Wt[(k0 + i + (i >= 8 ? 8 : 0)) * 64 + col];
        Wsw[f] = frag;
    }
    __syncthreads();

    const int lane = threadIdx.x & 31;
    const int wave = threadIdx.x >> 5;
    const int row0 = (blockIdx.x * 8 + wave) * 16;
    if (row0 >= nrows) return;                   // whole wave exits together

    const int  half = lane >> 4;                 // 0: lanes 0-15, 1: 16-31
    const int  l15  = lane & 15;
    const int  r    = row0 + l15;                // A row for this lane
    const bool rv   = (r < nrows);
    const int  ko   = half ? 8 : 0;

    v8f acc[4] = {};

#pragma unroll
    for (int kbb = 0; kbb < KB; ++kbb) {
        const int kb = kbb * 32;
        // --- A fragment: two contiguous 8-float runs -> 4x b128 loads ------
        const float* srcrow = (kb < W1)
            ? (A1 + (size_t)r * W1 + kb)
            : (A2 + (size_t)r * W2 + (kb - W1));
        v4f c0 = {}, c1 = {}, c2 = {}, c3 = {};
        if (rv) {
            c0 = *(const v4f*)(srcrow + ko);
            c1 = *(const v4f*)(srcrow + ko + 4);
            c2 = *(const v4f*)(srcrow + 16 + ko);
            c3 = *(const v4f*)(srcrow + 16 + ko + 4);
        }
        v16bf a;
#pragma unroll
        for (int i = 0; i < 4; ++i) {
            a[i]      = (__bf16)c0[i];
            a[4 + i]  = (__bf16)c1[i];
            a[8 + i]  = (__bf16)c2[i];
            a[12 + i] = (__bf16)c3[i];
        }
        // --- 4 column tiles: B fragment = one v16bf from LDS ---------------
#pragma unroll
        for (int ct = 0; ct < 4; ++ct) {
            v16bf b = Wsw[kbb * 128 + ct * 32 + lane];
            acc[ct] = __builtin_amdgcn_wmma_f32_16x16x32_bf16(
                false, a, false, b, (short)0, acc[ct], false, false);
        }
    }

    // --- epilogue: C/D layout -> lane = col, VGPR j = row -------------------
    const int rb = row0 + (half ? 8 : 0);
    if (row0 + 16 <= nrows) {
        // fast path: full tile, straight-line stores with immediate offsets
#pragma unroll
        for (int ct = 0; ct < 4; ++ct) {
            const int col = ct * 16 + l15;
            const float bs = bias ? bias[col] : 0.0f;
            float* p = Out + (size_t)rb * 64 + col;
#pragma unroll
            for (int j = 0; j < 8; ++j) {
                float v = acc[ct][j] + bs;
                if (RELU) v = fmaxf(v, 0.0f);
                p[(size_t)j * 64] = v;
            }
        }
    } else {
        // ragged tail tile: per-row guard
#pragma unroll
        for (int ct = 0; ct < 4; ++ct) {
            const int col = ct * 16 + l15;
            const float bs = bias ? bias[col] : 0.0f;
#pragma unroll
            for (int j = 0; j < 8; ++j) {
                int rr = rb + j;
                if (rr >= nrows) continue;
                float v = acc[ct][j] + bs;
                if (RELU) v = fmaxf(v, 0.0f);
                Out[(size_t)rr * 64 + col] = v;
            }
        }
    }
}

// ---------------------------------------------------------------------------
// Fold We[l] (16x64) against att_edge[l] (4x16)  ->  eM (16x4)
// ---------------------------------------------------------------------------
__global__ void fold_edge_att_kernel(const float* __restrict__ We_l,
                                     const float* __restrict__ ae_l,
                                     float* __restrict__ eM) {
    int t = threadIdx.x;
    if (t >= DE * HH) return;
    int k = t >> 2, h = t & 3;
    float s = 0.0f;
#pragma unroll
    for (int c = 0; c < CC; ++c) s += We_l[k * 64 + h * CC + c] * ae_l[h * CC + c];
    eM[k * HH + h] = s;
}

// a_src[n,h] = <xp[n,h,:], att_src[h,:]> ; same for dst
__global__ __launch_bounds__(256) void node_att_kernel(
    const float* __restrict__ xp, const float* __restrict__ as_l,
    const float* __restrict__ ad_l, float* __restrict__ a_src,
    float* __restrict__ a_dst) {
    size_t i = (size_t)blockIdx.x * blockDim.x + threadIdx.x;
    if (i >= (size_t)NN * HH) return;
    size_t n = i >> 2;
    int h = (int)(i & 3);
    const float* xr = xp + n * 64 + h * CC;
    float s1 = 0.0f, s2 = 0.0f;
#pragma unroll
    for (int c = 0; c < CC; ++c) { s1 += xr[c] * as_l[h * CC + c]; s2 += xr[c] * ad_l[h * CC + c]; }
    a_src[i] = s1; a_dst[i] = s2;
}

// a_edge[e,h] = eattr_f[e,:] @ eM  (self loops use loop_attr)
__global__ __launch_bounds__(256) void edge_att_kernel(
    const float* __restrict__ eattr, const float* __restrict__ loop_attr,
    const float* __restrict__ eM, float* __restrict__ a_edge) {
    int e = blockIdx.x * blockDim.x + threadIdx.x;
    if (e >= EF) return;
    const float* row = (e < EE) ? (eattr + (size_t)e * DE)
                                : (loop_attr + (size_t)(e - EE) * DE);
    float acc[HH] = {0.f, 0.f, 0.f, 0.f};
#pragma unroll
    for (int k = 0; k < DE; ++k) {
        float rk = row[k];
#pragma unroll
        for (int h = 0; h < HH; ++h) acc[h] += rk * eM[k * HH + h];
    }
#pragma unroll
    for (int h = 0; h < HH; ++h) a_edge[(size_t)e * HH + h] = acc[h];
}

// pass 1: alpha = leaky_relu(a_src[s]+a_dst[d]+a_edge) ; segment max over dst
__global__ __launch_bounds__(256) void alpha_max_kernel(
    const int* __restrict__ src, const int* __restrict__ dst,
    const float* __restrict__ a_src, const float* __restrict__ a_dst,
    const float* __restrict__ a_edge, float* __restrict__ alpha,
    float* __restrict__ mx) {
    int e = blockIdx.x * blockDim.x + threadIdx.x;
    if (e >= EF) return;
    int s, d;
    if (e < EE) { s = src[e]; d = dst[e]; } else { s = d = e - EE; }
#pragma unroll
    for (int h = 0; h < HH; ++h) {
        float v = a_src[(size_t)s * HH + h] + a_dst[(size_t)d * HH + h]
                + a_edge[(size_t)e * HH + h];
        v = (v > 0.0f) ? v : 0.2f * v;
        alpha[(size_t)e * HH + h] = v;
        atomicMaxFloat(&mx[(size_t)d * HH + h], v);
    }
}

// pass 2: ex = exp(alpha - mx[d]) ; segment sum over dst (alpha overwritten)
__global__ __launch_bounds__(256) void exp_sum_kernel(
    const int* __restrict__ dst, float* __restrict__ alpha,
    const float* __restrict__ mx, float* __restrict__ sm) {
    int e = blockIdx.x * blockDim.x + threadIdx.x;
    if (e >= EF) return;
    int d = (e < EE) ? dst[e] : (e - EE);
#pragma unroll
    for (int h = 0; h < HH; ++h) {
        float ex = __expf(alpha[(size_t)e * HH + h] - mx[(size_t)d * HH + h]);
        alpha[(size_t)e * HH + h] = ex;
        atomicAdd(&sm[(size_t)d * HH + h], ex);
    }
}

// pass 3: acc[d,c] += xp[s,c] * ex/sm[d]   (thread per (edge, channel))
__global__ __launch_bounds__(256) void scatter_kernel(
    const int* __restrict__ src, const int* __restrict__ dst,
    const float* __restrict__ alpha, const float* __restrict__ sm,
    const float* __restrict__ xp, float* __restrict__ acc) {
    size_t i = (size_t)blockIdx.x * blockDim.x + threadIdx.x;
    if (i >= (size_t)EF * 64) return;
    int e = (int)(i >> 6);
    int c = (int)(i & 63);
    int h = c >> 4;
    int s, d;
    if (e < EE) { s = src[e]; d = dst[e]; } else { s = d = e - EE; }
    float coef = alpha[(size_t)e * HH + h] / sm[(size_t)d * HH + h];
    atomicAdd(&acc[(size_t)d * 64 + c], xp[(size_t)s * 64 + c] * coef);
}

// ---------------------------------------------------------------------------
// Wave-per-row LayerNorm (+optional pre-bias, +optional ELU). wave32 shuffles.
// ---------------------------------------------------------------------------
__global__ __launch_bounds__(256) void ln_row_kernel(
    const float* __restrict__ in, const float* __restrict__ addb,
    const float* __restrict__ g, const float* __restrict__ b,
    float* __restrict__ out, int n, int do_elu) {
    int lane = threadIdx.x & 31;
    int wave = threadIdx.x >> 5;
    int node = blockIdx.x * 8 + wave;
    if (node >= n) return;
    const float* row = in + (size_t)node * 64;
    int c0 = 2 * lane, c1 = 2 * lane + 1;
    float x0 = row[c0] + (addb ? addb[c0] : 0.0f);
    float x1 = row[c1] + (addb ? addb[c1] : 0.0f);
    float s = x0 + x1;
#pragma unroll
    for (int m = 16; m >= 1; m >>= 1) s += __shfl_xor(s, m, 32);
    float mean = s * (1.0f / 64.0f);
    float d0 = x0 - mean, d1 = x1 - mean;
    float v = d0 * d0 + d1 * d1;
#pragma unroll
    for (int m = 16; m >= 1; m >>= 1) v += __shfl_xor(v, m, 32);
    float inv = rsqrtf(v * (1.0f / 64.0f) + 1e-5f);
    float y0 = d0 * inv * g[c0] + b[c0];
    float y1 = d1 * inv * g[c1] + b[c1];
    if (do_elu) {
        y0 = (y0 > 0.0f) ? y0 : (__expf(y0) - 1.0f);
        y1 = (y1 > 0.0f) ? y1 : (__expf(y1) - 1.0f);
    }
    out[(size_t)node * 64 + c0] = y0;
    out[(size_t)node * 64 + c1] = y1;
}

// ---------------------------------------------------------------------------
// Graph mean pooling
// ---------------------------------------------------------------------------
__global__ __launch_bounds__(256) void graph_pool_kernel(
    const float* __restrict__ gbuf, const int* __restrict__ batch,
    float* __restrict__ gsum, float* __restrict__ gcnt) {
    size_t i = (size_t)blockIdx.x * blockDim.x + threadIdx.x;
    if (i >= (size_t)NN * 64) return;
    int n = (int)(i >> 6);
    int c = (int)(i & 63);
    int bi = batch[n];
    atomicAdd(&gsum[(size_t)bi * 64 + c], gbuf[i]);
    if (c == 0) atomicAdd(&gcnt[bi], 1.0f);
}

__global__ void graph_div_kernel(const float* __restrict__ gsum,
                                 const float* __restrict__ gcnt,
                                 float* __restrict__ out) {
    int i = blockIdx.x * blockDim.x + threadIdx.x;
    if (i >= BB * 64) return;
    out[i] = gsum[i] / fmaxf(gcnt[i >> 6], 1.0f);
}

// ---------------------------------------------------------------------------
// Host-side orchestration
// ---------------------------------------------------------------------------
extern "C" void kernel_launch(void* const* d_in, const int* in_sizes, int n_in,
                              void* d_out, int out_size, void* d_ws, size_t ws_size,
                              hipStream_t stream) {
    (void)in_sizes; (void)n_in; (void)out_size; (void)ws_size;

    const float* x       = (const float*)d_in[0];
    const int*   eidx    = (const int*)  d_in[1];
    const float* eattr   = (const float*)d_in[2];
    const int*   batch   = (const int*)  d_in[3];
    const float* W_in    = (const float*)d_in[4];
    const float* b_in    = (const float*)d_in[5];
    const float* Wg      = (const float*)d_in[6];
    const float* att_src = (const float*)d_in[7];
    const float* att_dst = (const float*)d_in[8];
    const float* We      = (const float*)d_in[9];
    const float* att_edge= (const float*)d_in[10];
    const float* bg      = (const float*)d_in[11];
    const float* ln_g    = (const float*)d_in[12];
    const float* ln_b    = (const float*)d_in[13];
    const float* W_out   = (const float*)d_in[14];
    const float* b_out   = (const float*)d_in[15];
    const float* lnf_g   = (const float*)d_in[16];
    const float* lnf_b   = (const float*)d_in[17];
    const float* Wgr     = (const float*)d_in[18];
    const float* bgr     = (const float*)d_in[19];
    const float* lng_g   = (const float*)d_in[20];
    const float* lng_b   = (const float*)d_in[21];

    const int* src = eidx;
    const int* dst = eidx + EE;

    // workspace layout (floats)
    float* W = (float*)d_ws;
    size_t o = 0;
    float* deg       = W + o; o += NN;
    float* loop_attr = W + o; o += (size_t)NN * DE;
    float* h         = W + o; o += (size_t)NN * 64;
    float* xp        = W + o; o += (size_t)NN * 64;   // also reused as `tmp`
    float* a_src     = W + o; o += (size_t)NN * HH;
    float* a_dst     = W + o; o += (size_t)NN * HH;
    float* a_edge    = W + o; o += (size_t)EF * HH;
    float* alpha     = W + o; o += (size_t)EF * HH;   // reused as ex
    float* mx        = W + o; o += (size_t)NN * HH;
    float* sm        = W + o; o += (size_t)NN * HH;
    float* acc       = W + o; o += (size_t)NN * 64;   // also reused as gbuf
    float* gsum      = W + o; o += (size_t)BB * 64;
    float* gcnt      = W + o; o += BB;
    float* eM        = W + o; o += DE * HH;

    float* out_nodes = (float*)d_out;                   // [NN,64]
    float* out_graph = (float*)d_out + (size_t)NN * 64; // [BB,64]

    const int T = 256;
    const int gemm_blocks = CDIV(CDIV(NN, 16), 8);
    const float NEG_INF = -__builtin_huge_valf();

    // ---- self-loop edge-attr mean (shared by both layers) -----------------
    fill_kernel<<<CDIV(NN, T), T, 0, stream>>>(deg, NN, 0.0f);
    fill_kernel<<<CDIV((size_t)NN * DE, T), T, 0, stream>>>(loop_attr, (size_t)NN * DE, 0.0f);
    deg_loop_kernel<<<CDIV(EE, T), T, 0, stream>>>(dst, eattr, deg, loop_attr);
    loop_div_kernel<<<CDIV((size_t)NN * DE, T), T, 0, stream>>>(loop_attr, deg);

    // ---- node_in_fc: h = relu(x @ W_in + b_in) ----------------------------
    wmma_gemm_t<DIN, 0, 1><<<gemm_blocks, T, 0, stream>>>(x, nullptr, W_in, b_in, h, NN);

    // ---- GAT layers -------------------------------------------------------
    for (int l = 0; l < LL; ++l) {
        fold_edge_att_kernel<<<1, 64, 0, stream>>>(We + (size_t)l * DE * 64,
                                                   att_edge + (size_t)l * 64, eM);
        // xp = concat(x, h) @ Wg[l]      (gin never materialized)
        wmma_gemm_t<DIN, DEMB, 0><<<gemm_blocks, T, 0, stream>>>(
            x, h, Wg + (size_t)l * DGIN * 64, nullptr, xp, NN);
        node_att_kernel<<<CDIV((size_t)NN * HH, T), T, 0, stream>>>(
            xp, att_src + (size_t)l * 64, att_dst + (size_t)l * 64, a_src, a_dst);
        edge_att_kernel<<<CDIV(EF, T), T, 0, stream>>>(eattr, loop_attr, eM, a_edge);

        fill_kernel<<<CDIV((size_t)NN * HH, T), T, 0, stream>>>(mx, (size_t)NN * HH, NEG_INF);
        fill_kernel<<<CDIV((size_t)NN * HH, T), T, 0, stream>>>(sm, (size_t)NN * HH, 0.0f);
        fill_kernel<<<CDIV((size_t)NN * 64, T), T, 0, stream>>>(acc, (size_t)NN * 64, 0.0f);

        alpha_max_kernel<<<CDIV(EF, T), T, 0, stream>>>(src, dst, a_src, a_dst, a_edge, alpha, mx);
        exp_sum_kernel<<<CDIV(EF, T), T, 0, stream>>>(dst, alpha, mx, sm);
        scatter_kernel<<<CDIV((size_t)EF * 64, T), T, 0, stream>>>(src, dst, alpha, sm, xp, acc);

        // h = elu(LN(acc + bg[l]))
        ln_row_kernel<<<CDIV(NN, 8), T, 0, stream>>>(
            acc, bg + (size_t)l * 64, ln_g + (size_t)l * 64, ln_b + (size_t)l * 64, h, NN, 1);
    }

    // ---- node_emb = LN(relu(h @ W_out + b_out)) ---------------------------
    wmma_gemm_t<DEMB, 0, 1><<<gemm_blocks, T, 0, stream>>>(h, nullptr, W_out, b_out, xp, NN);
    ln_row_kernel<<<CDIV(NN, 8), T, 0, stream>>>(xp, nullptr, lnf_g, lnf_b, out_nodes, NN, 0);

    // ---- graph_emb = segment_mean(LN(relu(h @ Wgr + bgr))) ----------------
    wmma_gemm_t<DEMB, 0, 1><<<gemm_blocks, T, 0, stream>>>(h, nullptr, Wgr, bgr, xp, NN);
    ln_row_kernel<<<CDIV(NN, 8), T, 0, stream>>>(xp, nullptr, lng_g, lng_b, acc, NN, 0);
    fill_kernel<<<CDIV(BB * 64, T), T, 0, stream>>>(gsum, BB * 64, 0.0f);
    fill_kernel<<<1, T, 0, stream>>>(gcnt, BB, 0.0f);
    graph_pool_kernel<<<CDIV((size_t)NN * 64, T), T, 0, stream>>>(acc, batch, gsum, gcnt);
    graph_div_kernel<<<CDIV(BB * 64, T), T, 0, stream>>>(gsum, gcnt, out_graph);
}
